// EncoderRNN_5763846111973
// MI455X (gfx1250) — matrix operand order repaired and checked by reading, not compiled, not run on hardware
//
#include <hip/hip_runtime.h>
#include <math.h>

typedef __attribute__((ext_vector_type(16))) _Float16 v16h;
typedef __attribute__((ext_vector_type(8)))  float    v8f;
typedef __attribute__((ext_vector_type(4)))  float    f32x4;

#define HID  2048
#define EMBD 1024

// ---------------------------------------------------------------------------
// Kernel 0: gates[layer][dir][8192] = b_ih + b_hh  (fresh every call)
// ---------------------------------------------------------------------------
__global__ void init_gates_k(const float* __restrict__ bih0, const float* __restrict__ bhh0,
                             const float* __restrict__ bih1, const float* __restrict__ bhh1,
                             float* __restrict__ g0, float* __restrict__ g1) {
    int i = blockIdx.x * blockDim.x + threadIdx.x;   // 0..16383 (2 dirs x 8192)
    g0[i] = bih0[i] + bhh0[i];
    g1[i] = bih1[i] + bhh1[i];
}

// ---------------------------------------------------------------------------
// Kernel 1: x0 = emb[input_ids[0]]   (int64 id; LE low word is the value)
// ---------------------------------------------------------------------------
__global__ void gather_emb_k(const float* __restrict__ emb, const int* __restrict__ ids,
                             float* __restrict__ x0) {
    int i = blockIdx.x * blockDim.x + threadIdx.x;   // 0..1023
    long row = (long)ids[0];
    x0[i] = emb[row * EMBD + i];
}

// ---------------------------------------------------------------------------
// Split-K GEMV via v_wmma_f32_16x16x32_f16.
//   CX/CH : number of 32-wide K chunks for W_ih (x part) / W_hh (h part)
//   KSEG  : split-K factor; each wave handles CPW=(CX+CH)/KSEG chunks, 16 rows.
// Each wave's chunk range lies entirely in the x-region or entirely in the
// h-region (CX is a multiple of CPW), so the matrix/vector selection is
// hoisted out of the loop; the inner loop is pure pointer-increment.
// A: all 16 M-rows = current x chunk  -> every D row holds the 16 dots.
// B: 16 weight rows as 16 columns; lane (n, halfsel) holds 16 contiguous fp32.
// Partials combined with native f32 global atomics into bias-preloaded gates[].
// ---------------------------------------------------------------------------
template <int CX, int CH, int KSEG>
__global__ void gemv_gates_k(const float* __restrict__ Wih, const float* __restrict__ Whh,
                             const float* __restrict__ x,   const float* __restrict__ hbase,
                             float* __restrict__ gates) {
    constexpr int KX  = CX * 32;
    constexpr int KH  = CH * 32;
    constexpr int CPW = (CX + CH) / KSEG;            // chunks per wave (=16 here)
    static_assert(CX % CPW == 0, "wave chunk range must not straddle x/h boundary");

    const int wave    = blockIdx.x * (blockDim.x >> 5) + (threadIdx.x >> 5);
    const int lane    = threadIdx.x & 31;
    const int n       = lane & 15;                   // B column = weight row offset
    const int halfsel = lane >> 4;                   // K half selector within chunk

    const int kseg    = wave % KSEG;
    const int rowTile = (wave / KSEG) & 511;         // 512 tiles of 16 rows
    const int dir     = wave / (KSEG * 512);         // 0 = fwd, 1 = bwd
    const int rowBase = rowTile * 16;
    const int row     = rowBase + n;
    const int cBeg    = kseg * CPW;

    // Hoisted region selection (wave-uniform).
    const bool inX = (cBeg < CX);
    const float* vec;
    const float* Wr;                                 // this lane's weight stream
    if (inX) {
        const int kb0 = cBeg * 32;
        vec = x + kb0;
        Wr  = Wih + (size_t)dir * 8192 * KX + (size_t)row * KX + kb0 + halfsel * 16;
    } else {
        const int kb0 = (cBeg - CX) * 32;
        vec = hbase + dir * HID + kb0;
        Wr  = Whh + (size_t)dir * 8192 * KH + (size_t)row * KH + kb0 + halfsel * 16;
    }
    const float* av0 = vec + halfsel * 16;           // A elems 0..7  source
    const float* av1 = vec + halfsel * 16 + 8;       // A elems 8..15 source

    v8f acc = {};
#pragma unroll
    for (int c = 0; c < CPW; ++c) {
        // ---- A fragment: 16x32 f16, all rows = vec[c*32 .. c*32+31] ----
        // lanes 0-15 : elems 0..7 = K 0..7,   elems 8..15 = K 8..15
        // lanes 16-31: elems 0..7 = K 16..23, elems 8..15 = K 24..31
        f32x4 p0 = ((const f32x4*)(av0 + c * 32))[0];
        f32x4 p1 = ((const f32x4*)(av0 + c * 32))[1];
        f32x4 q0 = ((const f32x4*)(av1 + c * 32))[0];
        f32x4 q1 = ((const f32x4*)(av1 + c * 32))[1];
        v16h A;
#pragma unroll
        for (int i = 0; i < 4; ++i) {
            A[i]      = (_Float16)p0[i];
            A[4 + i]  = (_Float16)p1[i];
            A[8 + i]  = (_Float16)q0[i];
            A[12 + i] = (_Float16)q1[i];
        }

        // ---- B fragment: 32x16 f16; lane holds 16 contiguous fp32 (64B) ----
        const f32x4* w4 = (const f32x4*)(Wr + c * 32);
        f32x4 b0 = w4[0], b1 = w4[1], b2 = w4[2], b3 = w4[3];
        v16h B;
#pragma unroll
        for (int i = 0; i < 4; ++i) {
            B[i]      = (_Float16)b0[i];
            B[4 + i]  = (_Float16)b1[i];
            B[8 + i]  = (_Float16)b2[i];
            B[12 + i] = (_Float16)b3[i];
        }

        acc = __builtin_amdgcn_wmma_f32_16x16x32_f16(false, A, false, B,
                                                     (short)0, acc, false, false);
    }

    // All D rows identical (A rows broadcast) -> acc[0] on lanes 0..15 holds
    // the 16 partial dot products for rows rowBase..rowBase+15.
    if (lane < 16) {
        unsafeAtomicAdd(&gates[dir * 8192 + rowBase + n], acc[0]);
    }
}

// ---------------------------------------------------------------------------
// Pointwise LSTM cell (PyTorch gate order i,f,g,o). 2 dirs x 2048 = 4096 thr.
// c0l points at this layer's c0 slice; hn/cn at this layer's d_out slices.
// ---------------------------------------------------------------------------
__global__ void lstm_cell_k(const float* __restrict__ gates, const float* __restrict__ c0l,
                            float* __restrict__ hn, float* __restrict__ cn,
                            float* __restrict__ xnext, float* __restrict__ outbuf) {
    int t   = blockIdx.x * blockDim.x + threadIdx.x;   // 0..4095
    int dir = t >> 11;
    int j   = t & 2047;
    const float* g = gates + dir * 8192;
    float ig = g[j], fg = g[2048 + j], gg = g[4096 + j], og = g[6144 + j];
    float si = 1.0f / (1.0f + expf(-ig));
    float sf = 1.0f / (1.0f + expf(-fg));
    float so = 1.0f / (1.0f + expf(-og));
    float c2 = sf * c0l[dir * 2048 + j] + si * tanhf(gg);
    float h2 = so * tanhf(c2);
    hn[dir * 2048 + j] = h2;
    cn[dir * 2048 + j] = c2;
    if (xnext)  xnext[dir * 2048 + j]  = h2;  // concat(hf,hb) feeds next layer
    if (outbuf) outbuf[dir * 2048 + j] = h2;  // final output = concat(hf,hb)
}

// ---------------------------------------------------------------------------
extern "C" void kernel_launch(void* const* d_in, const int* in_sizes, int n_in,
                              void* d_out, int out_size, void* d_ws, size_t ws_size,
                              hipStream_t stream) {
    const int*   ids   = (const int*)d_in[0];
    const float* h0    = (const float*)d_in[1];     // [4,1,2048]
    const float* c0    = (const float*)d_in[2];     // [4,1,2048]
    const float* emb   = (const float*)d_in[3];     // [VOCAB,1024]
    const float* Wih0  = (const float*)d_in[4];     // [2,8192,1024]
    const float* Whh0  = (const float*)d_in[5];     // [2,8192,2048]
    const float* bih0  = (const float*)d_in[6];
    const float* bhh0  = (const float*)d_in[7];
    const float* Wih1  = (const float*)d_in[8];     // [2,8192,4096]
    const float* Whh1  = (const float*)d_in[9];     // [2,8192,2048]
    const float* bih1  = (const float*)d_in[10];
    const float* bhh1  = (const float*)d_in[11];

    float* out = (float*)d_out;          // [4096 output][8192 h_n][8192 c_n]
    float* hn  = out + 4096;
    float* cn  = out + 4096 + 8192;

    float* gates0 = (float*)d_ws;        // 16384 f32
    float* gates1 = gates0 + 16384;      // 16384 f32
    float* x0     = gates1 + 16384;      // 1024 f32 (embedded token)
    float* x1     = x0 + 1024;           // 4096 f32 (layer-1 input)

    // biases into gate accumulators (fresh each call -> replay-safe)
    init_gates_k<<<64, 256, 0, stream>>>(bih0, bhh0, bih1, bhh1, gates0, gates1);
    gather_emb_k<<<4, 256, 0, stream>>>(emb, ids, x0);

    // Layer 0: CX=32 (K=1024), CH=64 (K=2048), split-K 6 -> 6144 waves
    gemv_gates_k<32, 64, 6><<<768, 256, 0, stream>>>(Wih0, Whh0, x0, h0, gates0);
    lstm_cell_k<<<16, 256, 0, stream>>>(gates0, c0, hn, cn, x1, nullptr);

    // Layer 1: CX=128 (K=4096), CH=64 (K=2048), split-K 12 -> 12288 waves
    gemv_gates_k<128, 64, 12><<<1536, 256, 0, stream>>>(Wih1, Whh1, x1, h0 + 2 * HID, gates1);
    lstm_cell_k<<<16, 256, 0, stream>>>(gates1, c0 + 2 * HID, hn + 2 * HID, cn + 2 * HID,
                                        nullptr, out);
}
